// VectorQuantizerEMA_6021544149260
// MI455X (gfx1250) — matrix operand order repaired and checked by reading, not compiled, not run on hardware
//
#include <hip/hip_runtime.h>
#include <hip/hip_bf16.h>

// Problem constants (from reference setup_inputs)
#define F_DIM 8
#define B_DIM 16384
#define D_DIM 128
#define K_DIM 1024
#define DECAY_C 0.99f
#define EPS_C 1e-5f
#define CC_C 0.25f

typedef __attribute__((ext_vector_type(16))) __bf16 v16bf;
typedef __attribute__((ext_vector_type(8)))  float  v8f;

union BfVec {
    v16bf v;
    unsigned short s[16];
};

// ---- bf16 helpers (bit-exact, no reliance on __bf16 arithmetic) ----
__device__ __forceinline__ unsigned short f32_to_bf16(float f) {
    unsigned int u = __float_as_uint(f);
    unsigned int lsb = (u >> 16) & 1u;
    u += 0x7fffu + lsb;                 // round to nearest even
    return (unsigned short)(u >> 16);
}
__device__ __forceinline__ float bf16_to_f32(unsigned short h) {
    return __uint_as_float(((unsigned int)h) << 16);
}

// ---------------------------------------------------------------------------
// Kernel 1: transpose codebook to K-major bf16 hi/lo + column squared norms.
// w: [F, D, K] f32  ->  wThi/wTlo: [F, K, D] bf16, wsq: [F, K] f32
// Coalesced reads: consecutive threads = consecutive k at fixed d.
// ---------------------------------------------------------------------------
__global__ __launch_bounds__(256) void vq_prep(const float* __restrict__ w,
                                               unsigned short* __restrict__ wThi,
                                               unsigned short* __restrict__ wTlo,
                                               float* __restrict__ wsq) {
    const int f = blockIdx.y;
    const int k = blockIdx.x * 256 + threadIdx.x;
    const float* wf = w + (size_t)f * D_DIM * K_DIM;
    unsigned short* th = wThi + ((size_t)f * K_DIM + k) * D_DIM;
    unsigned short* tl = wTlo + ((size_t)f * K_DIM + k) * D_DIM;
    float s = 0.0f;
    for (int d = 0; d < D_DIM; ++d) {
        float v = wf[(size_t)d * K_DIM + k];
        s += v * v;
        unsigned short hb = f32_to_bf16(v);
        th[d] = hb;
        tl[d] = f32_to_bf16(v - bf16_to_f32(hb));
    }
    wsq[f * K_DIM + k] = s;
}

// ---------------------------------------------------------------------------
// Kernel 2: WMMA distance scoring + argmin.
// score(b,k) = ||w_k||^2 - 2 * x_b . w_k   (row-constant ||x||^2 dropped)
// x.w via split-bf16: 3x V_WMMA_F32_16X16X32_BF16 per 16x16x32 step.
// Block = 256 threads = 8 waves; wave owns rows [blk*128 + wid*16, +16).
// ---------------------------------------------------------------------------
__global__ __launch_bounds__(256) void vq_argmin(const float* __restrict__ x,
                                                 const unsigned short* __restrict__ wThi,
                                                 const unsigned short* __restrict__ wTlo,
                                                 const float* __restrict__ wsq,
                                                 int* __restrict__ enc) {
    const int f    = blockIdx.y;
    const int wid  = threadIdx.x >> 5;
    const int lane = threadIdx.x & 31;
    const int h    = lane >> 4;          // K-half of the 16x32 A / 32x16 B tile
    const int m    = lane & 15;          // row (A) / column (B) within tile
    const int rowBase = blockIdx.x * 128 + wid * 16;

    // ---- Load A fragments (x rows), split into bf16 hi/lo.
    // A 16x32 layout (contiguous, matching B): lane m (h=0): elems 0..15 -> K=d0+0..15,
    // h=1: elems 0..15 -> K=d0+16..31.
    BfVec ah[4], al[4];
    const float* xrow = x + ((size_t)f * B_DIM + rowBase + m) * D_DIM;
#pragma unroll
    for (int c = 0; c < 4; ++c) {
        const float* p = xrow + 32 * c + 16 * h;
#pragma unroll
        for (int e = 0; e < 8; ++e) {
            float v0 = p[e];
            unsigned short hb0 = f32_to_bf16(v0);
            ah[c].s[e] = hb0;
            al[c].s[e] = f32_to_bf16(v0 - bf16_to_f32(hb0));
            float v1 = p[8 + e];
            unsigned short hb1 = f32_to_bf16(v1);
            ah[c].s[8 + e] = hb1;
            al[c].s[8 + e] = f32_to_bf16(v1 - bf16_to_f32(hb1));
        }
    }

    float best[8];
    int   bidx[8];
#pragma unroll
    for (int r = 0; r < 8; ++r) { best[r] = 3.402823466e38f; bidx[r] = 0; }

    const unsigned short* whb = wThi + (size_t)f * K_DIM * D_DIM;
    const unsigned short* wlb = wTlo + (size_t)f * K_DIM * D_DIM;
    const float* wsqf = wsq + f * K_DIM;

    for (int k0 = 0; k0 < K_DIM; k0 += 16) {
        const int col = k0 + m;
        // B fragment: lane holds 16 consecutive d-values of column `col`
        // (lanes 0-15: K=d0+0..15, lanes 16-31: K=d0+16..31). Contiguous 32B.
        const unsigned short* ph = whb + (size_t)col * D_DIM + 16 * h;
        const unsigned short* pl = wlb + (size_t)col * D_DIM + 16 * h;

        v8f acc = {};
#pragma unroll
        for (int c = 0; c < 4; ++c) {
            v16bf bh = *reinterpret_cast<const v16bf*>(ph + 32 * c);
            v16bf bl = *reinterpret_cast<const v16bf*>(pl + 32 * c);
            acc = __builtin_amdgcn_wmma_f32_16x16x32_bf16(false, ah[c].v, false, bh,
                                                          (short)0, acc, false, false);
            acc = __builtin_amdgcn_wmma_f32_16x16x32_bf16(false, al[c].v, false, bh,
                                                          (short)0, acc, false, false);
            acc = __builtin_amdgcn_wmma_f32_16x16x32_bf16(false, ah[c].v, false, bl,
                                                          (short)0, acc, false, false);
        }

        const float wq = wsqf[col];
#pragma unroll
        for (int r = 0; r < 8; ++r) {
            float dsc = wq - 2.0f * acc[r];      // lane = column N, vgpr r = row
            bool lt = dsc < best[r];             // strict: keep lowest k on tie
            best[r] = lt ? dsc : best[r];
            bidx[r] = lt ? col : bidx[r];
        }
    }

    // Cross-lane argmin within each 16-lane half (C/D layout: lanes 0-15 hold
    // row r, lanes 16-31 hold row r+8 for accumulator VGPR r).
#pragma unroll
    for (int r = 0; r < 8; ++r) {
        float bv = best[r];
        int   bi = bidx[r];
#pragma unroll
        for (int off = 8; off >= 1; off >>= 1) {
            float ov = __shfl_xor(bv, off, 32);
            int   oi = __shfl_xor(bi, off, 32);
            if (ov < bv || (ov == bv && oi < bi)) { bv = ov; bi = oi; }
        }
        if (m == 0)
            enc[(size_t)f * B_DIM + rowBase + r + 8 * h] = bi;
    }
}

// ---------------------------------------------------------------------------
// Kernel 3: gather quantized vectors, loss partial sums, dw scatter, counts.
// One 128-thread block per (f, b) row; thread = dimension d.
// ---------------------------------------------------------------------------
__global__ __launch_bounds__(128) void vq_gather(const float* __restrict__ x,
                                                 const float* __restrict__ w,
                                                 const int* __restrict__ enc,
                                                 float* __restrict__ out,
                                                 float* __restrict__ dw,
                                                 float* __restrict__ counts,
                                                 float* __restrict__ lossAcc) {
    const int f = blockIdx.y;
    const int b = blockIdx.x;
    const int d = threadIdx.x;
    const int k = enc[(size_t)f * B_DIM + b];

    const size_t xi = ((size_t)f * B_DIM + b) * D_DIM + d;
    const float xv = x[xi];
    const float qv = w[((size_t)f * D_DIM + d) * K_DIM + k];
    out[xi] = qv;                                   // straight-through output
    atomicAdd(&dw[((size_t)f * D_DIM + d) * K_DIM + k], xv);

    float s = (qv - xv) * (qv - xv);
#pragma unroll
    for (int off = 16; off >= 1; off >>= 1) s += __shfl_down(s, off, 32);

    __shared__ float red[4];
    const int lane = threadIdx.x & 31, wid = threadIdx.x >> 5;
    if (lane == 0) red[wid] = s;
    __syncthreads();
    if (threadIdx.x == 0) {
        atomicAdd(lossAcc, red[0] + red[1] + red[2] + red[3]);
        atomicAdd(&counts[f * K_DIM + k], 1.0f);
    }
}

// ---------------------------------------------------------------------------
// Kernel 4: new_cs + normalization factors per f.
// ---------------------------------------------------------------------------
__global__ __launch_bounds__(256) void vq_stats(const float* __restrict__ ema_cs,
                                                const float* __restrict__ counts,
                                                float* __restrict__ out_newcs,
                                                float* __restrict__ norm) {
    const int f = blockIdx.x;
    float cs[4];
    float local = 0.0f;
#pragma unroll
    for (int j = 0; j < 4; ++j) {
        const int k = threadIdx.x + 256 * j;
        float c = DECAY_C * ema_cs[f * K_DIM + k] + (1.0f - DECAY_C) * counts[f * K_DIM + k];
        out_newcs[f * K_DIM + k] = c;
        cs[j] = c;
        local += c;
    }
#pragma unroll
    for (int off = 16; off >= 1; off >>= 1) local += __shfl_down(local, off, 32);

    __shared__ float red[8];
    __shared__ float nsh;
    const int lane = threadIdx.x & 31, wid = threadIdx.x >> 5;
    if (lane == 0) red[wid] = local;
    __syncthreads();
    if (threadIdx.x == 0) {
        float n = 0.0f;
#pragma unroll
        for (int i = 0; i < 8; ++i) n += red[i];
        nsh = n;
    }
    __syncthreads();
    const float n = nsh;
#pragma unroll
    for (int j = 0; j < 4; ++j) {
        const int k = threadIdx.x + 256 * j;
        norm[f * K_DIM + k] = (cs[j] + EPS_C) / (n + K_DIM * EPS_C) * n;
    }
}

// ---------------------------------------------------------------------------
// Kernel 5: new_ema_w, new_w, and the scalar loss.
// ---------------------------------------------------------------------------
__global__ __launch_bounds__(256) void vq_final(const float* __restrict__ ema_w,
                                                const float* __restrict__ dw,
                                                const float* __restrict__ norm,
                                                float* __restrict__ out_neww,
                                                float* __restrict__ out_emaw,
                                                const float* __restrict__ lossAcc,
                                                float* __restrict__ out_loss) {
    const int i = blockIdx.x * 256 + threadIdx.x;      // i < F*D*K exactly
    const int k = i & (K_DIM - 1);
    const int f = i >> 17;                             // D*K = 2^17
    const float ne = DECAY_C * ema_w[i] + (1.0f - DECAY_C) * dw[i];
    out_emaw[i] = ne;
    out_neww[i] = ne / norm[f * K_DIM + k];
    if (i == 0)
        out_loss[0] = CC_C * lossAcc[0] / (float)((size_t)F_DIM * B_DIM * D_DIM);
}

// ---------------------------------------------------------------------------
extern "C" void kernel_launch(void* const* d_in, const int* in_sizes, int n_in,
                              void* d_out, int out_size, void* d_ws, size_t ws_size,
                              hipStream_t stream) {
    const float* x      = (const float*)d_in[0];   // [F,B,D]
    const float* w      = (const float*)d_in[1];   // [F,D,K]
    const float* ema_cs = (const float*)d_in[2];   // [F,K]
    const float* ema_w  = (const float*)d_in[3];   // [F,D,K]

    float* out = (float*)d_out;
    const size_t N_OUT  = (size_t)F_DIM * B_DIM * D_DIM;   // 16777216
    const size_t N_WDK  = (size_t)F_DIM * D_DIM * K_DIM;   // 1048576
    const size_t N_FK   = (size_t)F_DIM * K_DIM;           // 8192
    float* out_output = out;
    float* out_loss   = out + N_OUT;
    float* out_neww   = out + N_OUT + 1;
    float* out_newcs  = out + N_OUT + 1 + N_WDK;
    float* out_emaw   = out + N_OUT + 1 + N_WDK + N_FK;

    // Workspace carve-up (all 256B aligned)
    char* ws = (char*)d_ws;
    unsigned short* wThi = (unsigned short*)(ws + 0);            // 2 MiB
    unsigned short* wTlo = (unsigned short*)(ws + 2097152);      // 2 MiB
    float* wsq    = (float*)(ws + 4194304);                      // 32 KiB
    int*   enc    = (int*)  (ws + 4227072);                      // 512 KiB
    float* counts = (float*)(ws + 4751360);                      // 32 KiB
    float* dw     = (float*)(ws + 4784128);                      // 4 MiB
    float* norm   = (float*)(ws + 8978432);                      // 32 KiB
    float* lossAcc= (float*)(ws + 9011200);                      // 256 B

    // Zero the accumulators every call (harness poisons once, not per replay).
    hipMemsetAsync(counts, 0, N_FK * sizeof(float), stream);
    hipMemsetAsync(dw, 0, N_WDK * sizeof(float), stream);
    hipMemsetAsync(lossAcc, 0, 256, stream);

    vq_prep<<<dim3(K_DIM / 256, F_DIM), 256, 0, stream>>>(w, wThi, wTlo, wsq);
    vq_argmin<<<dim3(B_DIM / 128, F_DIM), 256, 0, stream>>>(x, wThi, wTlo, wsq, enc);
    vq_gather<<<dim3(B_DIM, F_DIM), 128, 0, stream>>>(x, w, enc, out_output, dw, counts, lossAcc);
    vq_stats<<<dim3(F_DIM), 256, 0, stream>>>(ema_cs, counts, out_newcs, norm);
    vq_final<<<dim3((unsigned)(N_WDK / 256)), 256, 0, stream>>>(ema_w, dw, norm, out_neww,
                                                                out_emaw, lossAcc, out_loss);
}
